// RC_STML_21114059227769
// MI455X (gfx1250) — matrix operand order repaired and checked by reading, not compiled, not run on hardware
//
#include <hip/hip_runtime.h>
#include <math.h>

// ---------------------------------------------------------------------------
// STML relational loss for MI455X (gfx1250, wave32, WMMA + async-to-LDS).
//
// Heavy work = two 4096x4096x512 Gram GEMMs done with
// v_wmma_f32_16x16x32_f16 (f16 in, f32 accumulate), with tiles staged in
// LDS via GLOBAL_LOAD_ASYNC_TO_LDS_B128 (ASYNCcnt-tracked, double
// buffered). Everything downstream (top-k, mutual-kNN graph, W_C) is
// sparse (<=50 nnz/row) and handled by cheap scalar kernels. Working set
// (2x64MB Gram + 8MB f16 embeddings) fits in the 192MB L2.
// ---------------------------------------------------------------------------

static constexpr int kN    = 4096;
static constexpr int kD    = 512;
static constexpr int kTopK = 10;
static constexpr int kHalf = 5;

typedef __attribute__((ext_vector_type(16))) _Float16 v16h;
typedef __attribute__((ext_vector_type(8)))  _Float16 v8h;
typedef __attribute__((ext_vector_type(8)))  float    v8f;

__device__ __forceinline__ void async_load_b128(unsigned lds_off,
                                                const void* gaddr) {
  asm volatile("global_load_async_to_lds_b128 %0, %1, off"
               :: "v"(lds_off),
                  "v"((unsigned long long)(size_t)gaddr)
               : "memory");
}

template <int NWait>
__device__ __forceinline__ void wait_asynccnt() {
#if __has_builtin(__builtin_amdgcn_s_wait_asynccnt)
  __builtin_amdgcn_s_wait_asynccnt((short)NWait);
#else
  asm volatile("s_wait_asynccnt %0" :: "i"(NWait) : "memory");
#endif
}

// ---------------------------------------------------------------------------
// 1) Row L2-normalize fp32 -> f16 (one block per row)
// ---------------------------------------------------------------------------
__global__ __launch_bounds__(128) void normalize_kernel(
    const float* __restrict__ in, _Float16* __restrict__ out) {
  __shared__ float red[128];
  const int row = blockIdx.x;
  const float* p = in + (size_t)row * kD;
  float s = 0.f;
  for (int e = threadIdx.x; e < kD; e += 128) { float v = p[e]; s += v * v; }
  red[threadIdx.x] = s;
  __syncthreads();
  for (int o = 64; o > 0; o >>= 1) {
    if (threadIdx.x < o) red[threadIdx.x] += red[threadIdx.x + o];
    __syncthreads();
  }
  const float scale = 1.0f / fmaxf(sqrtf(red[0]), 1e-12f);
  _Float16* q = out + (size_t)row * kD;
  for (int e = threadIdx.x; e < kD; e += 128)
    q[e] = (_Float16)(p[e] * scale);
}

// ---------------------------------------------------------------------------
// 2) Gram matrix C = A * A^T via WMMA f32_16x16x32_f16.
//    Block: 256 threads = 8 waves; block tile 128 rows x 64 cols.
//    K-tiles of 64 staged in LDS with GLOBAL_LOAD_ASYNC_TO_LDS_B128,
//    double-buffered (2-deep pipeline, s_wait_asynccnt retire).
//    Rows padded to 72 halves (144B) to spread LDS banks.
// ---------------------------------------------------------------------------
static constexpr int kKT  = 64;  // K-tile
static constexpr int kPad = 72;  // padded row length in halves

__global__ __launch_bounds__(256) void gram_kernel(
    const _Float16* __restrict__ A, float* __restrict__ C) {
  __shared__ __align__(32) _Float16 Asm[2][128 * kPad];
  __shared__ __align__(32) _Float16 Bsm[2][64 * kPad];

  const int tid  = threadIdx.x;
  const int lane = tid & 31;
  const int wave = tid >> 5;
  const int hi   = lane >> 4;    // lane half (0: lanes 0-15, 1: lanes 16-31)
  const int l15  = lane & 15;
  const int rowBase = blockIdx.y * 128;
  const int colBase = blockIdx.x * 64;

  v8f acc[4];
#pragma unroll
  for (int t = 0; t < 4; ++t) acc[t] = (v8f){};

  // Issue one K-tile (A: 128x64 f16 = 1024 16B chunks -> 4/thread;
  //                   B:  64x64 f16 =  512 16B chunks -> 2/thread)
  auto issue_tile = [&](int kb, int buf) {
#pragma unroll
    for (int q = 0; q < 4; ++q) {
      const int chunk = tid * 4 + q;
      const int row   = chunk >> 3;          // 8 chunks per 64-half row
      const int ko    = (chunk & 7) * 8;     // half offset within row
      const _Float16* g = A + (size_t)(rowBase + row) * kD + kb + ko;
      async_load_b128((unsigned)(size_t)&Asm[buf][row * kPad + ko], g);
    }
#pragma unroll
    for (int q = 0; q < 2; ++q) {
      const int chunk = tid * 2 + q;
      const int row   = chunk >> 3;
      const int ko    = (chunk & 7) * 8;
      const _Float16* g = A + (size_t)(colBase + row) * kD + kb + ko;
      async_load_b128((unsigned)(size_t)&Bsm[buf][row * kPad + ko], g);
    }
  };

  issue_tile(0, 0);

  const int nTiles = kD / kKT;  // 8
  for (int tt = 0; tt < nTiles; ++tt) {
    const int buf = tt & 1;
    if (tt + 1 < nTiles) {
      issue_tile((tt + 1) * kKT, buf ^ 1);
      wait_asynccnt<6>();   // retire current tile (6 older ops), keep 6 inflight
    } else {
      wait_asynccnt<0>();
    }
    __syncthreads();

    const _Float16* arow = &Asm[buf][(wave * 16 + l15) * kPad];
#pragma unroll
    for (int ks = 0; ks < kKT; ks += 32) {
      // A fragment (16x32 f16): elems 0..7 <- K=ks+8*hi+0..7,
      //                         elems 8..15 <- K=ks+16+8*hi+0..7
      v8h alo = *(const v8h*)(arow + ks + 8 * hi);
      v8h ahi = *(const v8h*)(arow + ks + 16 + 8 * hi);
      v16h a;
#pragma unroll
      for (int e = 0; e < 8; ++e) { a[e] = alo[e]; a[8 + e] = ahi[e]; }

#pragma unroll
      for (int t = 0; t < 4; ++t) {
        // B fragment (32x16 f16): elems 0..7 <- K=ks+8*hi+0..7, elems 8..15 <- K=ks+16+8*hi+0..7 of row `col`
        const _Float16* brow = &Bsm[buf][(t * 16 + l15) * kPad];
        v8h b0 = *(const v8h*)(brow + ks + 8 * hi);
        v8h b1 = *(const v8h*)(brow + ks + 16 + 8 * hi);
        v16h b;
#pragma unroll
        for (int e = 0; e < 8; ++e) { b[e] = b0[e]; b[8 + e] = b1[e]; }
        acc[t] = __builtin_amdgcn_wmma_f32_16x16x32_f16(
            /*neg_a=*/false, a, /*neg_b=*/false, b,
            /*c_mod=*/(short)0, acc[t], /*reuse_a=*/false, /*reuse_b=*/false);
      }
    }
    __syncthreads();
  }

  // C/D layout: VGPR r, lane -> element (M = 8*hi + r, N = lane&15)
#pragma unroll
  for (int t = 0; t < 4; ++t)
#pragma unroll
    for (int r = 0; r < 8; ++r)
      C[(size_t)(rowBase + wave * 16 + 8 * hi + r) * kN +
        colBase + t * 16 + l15] = acc[t][r];
}

// ---------------------------------------------------------------------------
// 3) Row sums of S_dist = sqrt(max(2-2g, 0))  (one block per row)
// ---------------------------------------------------------------------------
__global__ __launch_bounds__(256) void srow_kernel(
    const float* __restrict__ Gs, float* __restrict__ srow) {
  __shared__ float red[256];
  const int row = blockIdx.x;
  const float* p = Gs + (size_t)row * kN;
  float s = 0.f;
  for (int j = threadIdx.x; j < kN; j += 256)
    s += sqrtf(fmaxf(2.f - 2.f * p[j], 0.f));
  red[threadIdx.x] = s;
  __syncthreads();
  for (int o = 128; o > 0; o >>= 1) {
    if (threadIdx.x < o) red[threadIdx.x] += red[threadIdx.x + o];
    __syncthreads();
  }
  if (threadIdx.x == 0) srow[row] = red[0];
}

// ---------------------------------------------------------------------------
// 4) Per-row top-10 of W_P_copy (wave per row; exp is monotone in g, ties
//    broken by lower index to mimic jax.lax.top_k).
// ---------------------------------------------------------------------------
__global__ __launch_bounds__(256) void topk_kernel(
    const float* __restrict__ Gt, const int* __restrict__ idx,
    int* __restrict__ topk) {
  const int lane = threadIdx.x & 31;
  const int row  = blockIdx.x * 8 + (threadIdx.x >> 5);
  const int myid = idx[row];

  float bv[kTopK];
  int   bc[kTopK];
#pragma unroll
  for (int t = 0; t < kTopK; ++t) { bv[t] = -1e30f; bc[t] = 0x7fffffff; }

  const float* grow = Gt + (size_t)row * kN;
  for (int j = lane; j < kN; j += 32) {
    float v = (idx[j] == myid)
                  ? 1.0f
                  : __expf(-fmaxf(2.f - 2.f * grow[j], 0.f));
    if (v > bv[kTopK - 1] || (v == bv[kTopK - 1] && j < bc[kTopK - 1])) {
      int p = kTopK - 1;
      while (p > 0 && (v > bv[p - 1] || (v == bv[p - 1] && j < bc[p - 1]))) {
        bv[p] = bv[p - 1]; bc[p] = bc[p - 1]; --p;
      }
      bv[p] = v; bc[p] = j;
    }
  }

  // Merge 32 per-lane sorted lists: 10 rounds of wave-wide argmax.
  int h = 0;
  for (int t = 0; t < kTopK; ++t) {
    float v = (h < kTopK) ? bv[h] : -1e30f;
    int   c = (h < kTopK) ? bc[h] : 0x7fffffff;
    for (int off = 16; off > 0; off >>= 1) {
      float ov = __shfl_xor(v, off, 32);
      int   oc = __shfl_xor(c, off, 32);
      if (ov > v || (ov == v && oc < c)) { v = ov; c = oc; }
    }
    if (h < kTopK && c == bc[h]) ++h;  // columns are unique per lane
    if (lane == 0) topk[row * kTopK + t] = c;
  }
}

// ---------------------------------------------------------------------------
// 5) Mutual-kNN adjacency V (<=10 nnz/row)
// ---------------------------------------------------------------------------
__global__ void mutual_kernel(const int* __restrict__ topk,
                              int* __restrict__ vcols,
                              int* __restrict__ vcnt) {
  const int i = blockIdx.x * blockDim.x + threadIdx.x;
  if (i >= kN) return;
  int mine[kTopK];
#pragma unroll
  for (int s = 0; s < kTopK; ++s) mine[s] = topk[i * kTopK + s];
  int cnt = 0;
  for (int s = 0; s < kTopK; ++s) {
    const int j = mine[s];
    bool mut = false;
    for (int q = 0; q < kTopK; ++q)
      if (topk[j * kTopK + q] == i) mut = true;
    if (mut) vcols[i * kTopK + cnt++] = j;
  }
  vcnt[i] = cnt;
}

// ---------------------------------------------------------------------------
// 6) Sparse W_C_tilda values: VV(i,j)/max(rowcnt_i,1) at V>0 positions only
//    (replaces the dense 4096^3 V*V^T GEMM with 10x10 set intersections)
// ---------------------------------------------------------------------------
__global__ void vv_kernel(const int* __restrict__ vcols,
                          const int* __restrict__ vcnt,
                          float* __restrict__ tvals) {
  const int i = blockIdx.x * blockDim.x + threadIdx.x;
  if (i >= kN) return;
  const int ci = vcnt[i];
  int mine[kTopK];
  for (int m = 0; m < ci; ++m) mine[m] = vcols[i * kTopK + m];
  const float inv = 1.0f / fmaxf((float)ci, 1.0f);
  for (int m = 0; m < ci; ++m) {
    const int j  = mine[m];
    const int cj = vcnt[j];
    int common = 0;
    for (int a = 0; a < ci; ++a)
      for (int b = 0; b < cj; ++b)
        if (mine[a] == vcols[j * kTopK + b]) ++common;
    tvals[i * kTopK + m] = (float)common * inv;
  }
}

// ---------------------------------------------------------------------------
// 7) Accumulator init (graph-capture-safe zeroing)
// ---------------------------------------------------------------------------
__global__ void zero_accs(float* accs) { accs[0] = 0.f; accs[1] = 0.f; }

// ---------------------------------------------------------------------------
// 8) Dense part: sum_{i!=j} [ r^2 + 0.5*(d^2-r^2)*W_P ], d/r/W_P rebuilt
//    on the fly from the Gram matrices and row sums.
// ---------------------------------------------------------------------------
__global__ __launch_bounds__(256) void dense_loss_kernel(
    const float* __restrict__ Gs, const float* __restrict__ Gt,
    const float* __restrict__ srow, float* __restrict__ accs) {
  __shared__ float red[256];
  const long long total  = (long long)kN * kN;
  const long long stride = (long long)gridDim.x * blockDim.x;
  float acc = 0.f;
  for (long long p = (long long)blockIdx.x * blockDim.x + threadIdx.x;
       p < total; p += stride) {
    const int i = (int)(p >> 12);
    const int j = (int)(p & (kN - 1));
    if (i == j) continue;
    const float sd = sqrtf(fmaxf(2.f - 2.f * Gs[p], 0.f));
    const float d  = sd * ((float)kN / srow[i]);
    const float r  = fmaxf(1.f - d, 0.f);
    const float f  = d * d - r * r;
    const float wp = __expf(-fmaxf(2.f - 2.f * Gt[p], 0.f));
    acc += r * r + 0.5f * f * wp;
  }
  red[threadIdx.x] = acc;
  __syncthreads();
  for (int o = 128; o > 0; o >>= 1) {
    if (threadIdx.x < o) red[threadIdx.x] += red[threadIdx.x + o];
    __syncthreads();
  }
  if (threadIdx.x == 0) atomicAdd(&accs[0], red[0]);
}

// ---------------------------------------------------------------------------
// 9) Sparse W_C correction: sum_{i!=j} hat_ij*(f_ij+f_ji)/4
// ---------------------------------------------------------------------------
__global__ void sparse_loss_kernel(
    const int* __restrict__ topk, const int* __restrict__ vcols,
    const int* __restrict__ vcnt, const float* __restrict__ tvals,
    const float* __restrict__ Gs, const float* __restrict__ srow,
    float* __restrict__ accs) {
  const int i = blockIdx.x * blockDim.x + threadIdx.x;
  if (i >= kN) return;
  const float invm_i = (float)kN / srow[i];
  float acc = 0.f;
  for (int h = 0; h < kHalf; ++h) {
    const int k  = topk[i * kTopK + h];
    const int ck = vcnt[k];
    for (int m = 0; m < ck; ++m) {
      const int j = vcols[k * kTopK + m];
      if (j == i) continue;
      const float hat = tvals[k * kTopK + m] * (1.0f / kHalf);
      const float sd  = sqrtf(fmaxf(2.f - 2.f * Gs[(size_t)i * kN + j], 0.f));
      const float d1 = sd * invm_i;
      const float r1 = fmaxf(1.f - d1, 0.f);
      const float f1 = d1 * d1 - r1 * r1;
      const float d2 = sd * ((float)kN / srow[j]);
      const float r2 = fmaxf(1.f - d2, 0.f);
      const float f2 = d2 * d2 - r2 * r2;
      acc += hat * (f1 + f2) * 0.25f;
    }
  }
  atomicAdd(&accs[1], acc);
}

__global__ void finalize_kernel(const float* __restrict__ accs,
                                float* __restrict__ out) {
  out[0] = (accs[0] + accs[1]) / ((float)kN * (float)(kN - 1));
}

// ---------------------------------------------------------------------------
// Host-side orchestration
// ---------------------------------------------------------------------------
extern "C" void kernel_launch(void* const* d_in, const int* in_sizes, int n_in,
                              void* d_out, int out_size, void* d_ws,
                              size_t ws_size, hipStream_t stream) {
  (void)in_sizes; (void)n_in; (void)out_size; (void)ws_size;

  const float* s_emb = (const float*)d_in[0];
  const float* t_emb = (const float*)d_in[1];
  const int*   idx   = (const int*)d_in[2];
  float*       out   = (float*)d_out;

  char* w = (char*)d_ws;
  size_t off = 0;
  _Float16* sbf = (_Float16*)(w + off); off += (size_t)kN * kD * 2;   // 4 MB
  _Float16* tbf = (_Float16*)(w + off); off += (size_t)kN * kD * 2;   // 4 MB
  float* Gs   = (float*)(w + off); off += (size_t)kN * kN * 4;        // 64 MB
  float* Gt   = (float*)(w + off); off += (size_t)kN * kN * 4;        // 64 MB
  float* srow = (float*)(w + off); off += (size_t)kN * 4;
  int* topk   = (int*)(w + off);   off += (size_t)kN * kTopK * 4;
  int* vcols  = (int*)(w + off);   off += (size_t)kN * kTopK * 4;
  int* vcnt   = (int*)(w + off);   off += (size_t)kN * 4;
  float* tvals = (float*)(w + off); off += (size_t)kN * kTopK * 4;
  float* accs = (float*)(w + off);  off += 2 * 4;

  // 1) normalize both embeddings to f16
  normalize_kernel<<<kN, 128, 0, stream>>>(s_emb, sbf);
  normalize_kernel<<<kN, 128, 0, stream>>>(t_emb, tbf);

  // 2) Gram matrices via WMMA (LDS-staged, async double-buffered)
  dim3 gg(kN / 64, kN / 128);
  gram_kernel<<<gg, 256, 0, stream>>>(sbf, Gs);
  gram_kernel<<<gg, 256, 0, stream>>>(tbf, Gt);

  // 3) row sums of S_dist
  srow_kernel<<<kN, 256, 0, stream>>>(Gs, srow);

  // 4) top-10 per row of W_P_copy
  topk_kernel<<<kN / 8, 256, 0, stream>>>(Gt, idx, topk);

  // 5-6) mutual kNN graph + sparse W_C_tilda
  mutual_kernel<<<kN / 256, 256, 0, stream>>>(topk, vcols, vcnt);
  vv_kernel<<<kN / 256, 256, 0, stream>>>(vcols, vcnt, tvals);

  // 7-9) loss reduction
  zero_accs<<<1, 1, 0, stream>>>(accs);
  dense_loss_kernel<<<2048, 256, 0, stream>>>(Gs, Gt, srow, accs);
  sparse_loss_kernel<<<kN / 256, 256, 0, stream>>>(topk, vcols, vcnt, tvals,
                                                   Gs, srow, accs);
  finalize_kernel<<<1, 1, 0, stream>>>(accs, out);
}
